// MultiModalIntegrationGNN_5866925326769
// MI455X (gfx1250) — matrix operand order repaired and checked by reading, not compiled, not run on hardware
//
#include <hip/hip_runtime.h>

typedef __attribute__((ext_vector_type(16))) _Float16 v16h;
typedef __attribute__((ext_vector_type(8)))  _Float16 v8h;
typedef __attribute__((ext_vector_type(8)))  float    v8f;

#define NN 50000
#define NE 600000

__device__ __forceinline__ v8f vzero8() {
  v8f z;
#pragma unroll
  for (int i = 0; i < 8; ++i) z[i] = 0.0f;
  return z;
}

__device__ __forceinline__ v8h cvt8(const float4 a, const float4 b) {
  v8h h;
  h[0] = (_Float16)a.x; h[1] = (_Float16)a.y; h[2] = (_Float16)a.z; h[3] = (_Float16)a.w;
  h[4] = (_Float16)b.x; h[5] = (_Float16)b.y; h[6] = (_Float16)b.z; h[7] = (_Float16)b.w;
  return h;
}

// A fragment: 16x32 f16 tile from LDS. Per ISA layout:
// lanes 0-15: M=lane, halves 0-7 -> K 0-7, halves 8-15 -> K 16-23
// lanes 16-31: M=lane-16, halves 0-7 -> K 8-15, halves 8-15 -> K 24-31
__device__ __forceinline__ v16h lds_a_frag(const _Float16* tile, int rowStride, int lane) {
  const int m = lane & 15;
  const int koff = (lane >> 4) << 3;
  const _Float16* p = tile + m * rowStride + koff;
  v8h lo = *(const v8h*)(p);
  v8h hi = *(const v8h*)(p + 16);
  v16h a;
#pragma unroll
  for (int i = 0; i < 8; ++i) { a[i] = lo[i]; a[i + 8] = hi[i]; }
  return a;
}

// B fragment: 32x16 f16 from transposed weights Wt[n][k].
// lanes 0-15 hold K 0-7,16-23 of the chunk, lanes 16-31 hold K 8-15,24-31; N = lane&15.
__device__ __forceinline__ v16h lds_b_frag(const _Float16* colPtr, int lane) {
  const int kb = (lane >> 4) << 3;
  const _Float16* p = colPtr + kb;
  v8h lo = *(const v8h*)(p);
  v8h hi = *(const v8h*)(p + 16);
  v16h b;
#pragma unroll
  for (int i = 0; i < 8; ++i) { b[i] = lo[i]; b[i + 8] = hi[i]; }
  return b;
}

// One 16-row x 64-col x 32-K step: load A + all 4 B fragments first (single
// dscnt wait), then issue 4 back-to-back v_wmma so they can co-execute with
// surrounding VALU staging work.
__device__ __forceinline__ void wmma_row(const _Float16* Atile, int lda, int kc,
                                         const _Float16* Wt, int ldw,
                                         int lane, v8f acc[4]) {
  const int n0 = lane & 15;
  const v16h a  = lds_a_frag(Atile + kc, lda, lane);
  const v16h b0 = lds_b_frag(&Wt[(n0 +  0) * ldw + kc], lane);
  const v16h b1 = lds_b_frag(&Wt[(n0 + 16) * ldw + kc], lane);
  const v16h b2 = lds_b_frag(&Wt[(n0 + 32) * ldw + kc], lane);
  const v16h b3 = lds_b_frag(&Wt[(n0 + 48) * ldw + kc], lane);
  acc[0] = __builtin_amdgcn_wmma_f32_16x16x32_f16(false, a, false, b0, (short)0, acc[0], false, false);
  acc[1] = __builtin_amdgcn_wmma_f32_16x16x32_f16(false, a, false, b1, (short)0, acc[1], false, false);
  acc[2] = __builtin_amdgcn_wmma_f32_16x16x32_f16(false, a, false, b2, (short)0, acc[2], false, false);
  acc[3] = __builtin_amdgcn_wmma_f32_16x16x32_f16(false, a, false, b3, (short)0, acc[3], false, false);
}

// Stage W[rows][64] row-major (f32, global) -> Wt[64][ldWt] transposed (f16, LDS)
__device__ __forceinline__ void stage_wt(_Float16* Wt, int ldWt,
                                         const float* __restrict__ W,
                                         int rows, int tid) {
  for (int g = tid; g < rows * 8; g += 256) {
    const int k  = g >> 3;
    const int n0 = (g & 7) * 8;
    const float4 a = *(const float4*)(W + (long)k * 64 + n0);
    const float4 b = *(const float4*)(W + (long)k * 64 + n0 + 4);
    Wt[(n0 + 0) * ldWt + k] = (_Float16)a.x;
    Wt[(n0 + 1) * ldWt + k] = (_Float16)a.y;
    Wt[(n0 + 2) * ldWt + k] = (_Float16)a.z;
    Wt[(n0 + 3) * ldWt + k] = (_Float16)a.w;
    Wt[(n0 + 4) * ldWt + k] = (_Float16)b.x;
    Wt[(n0 + 5) * ldWt + k] = (_Float16)b.y;
    Wt[(n0 + 6) * ldWt + k] = (_Float16)b.z;
    Wt[(n0 + 7) * ldWt + k] = (_Float16)b.w;
  }
}

struct XRegs { float4 f[4]; };

// Issue global loads for one 32-wide K chunk (16 floats per thread) into regs.
// Chunk-uniform fast path (chunk wholly inside X1 or X2; other chunks take the per-element path).
__device__ __forceinline__ void load_x_chunk(XRegs& r,
    const float* __restrict__ X1, int K1,
    const float* __restrict__ X2, int K2, int K,
    int rg, bool rv, int k0, int kc) {
#pragma unroll
  for (int i = 0; i < 4; ++i) r.f[i] = make_float4(0.f, 0.f, 0.f, 0.f);
  const bool fast1 = (kc + 32 <= K1);
  const bool fast2 = (kc >= K1) && (kc + 32 <= K);
  if (fast1 || fast2) {
    const float* Xp = fast1 ? X1 : X2;
    const int    ld = fast1 ? K1 : K2;
    const long  off = (long)rg * ld + (fast1 ? (kc + k0) : (kc - K1 + k0));
    if (rv) {
      const float4* gp = (const float4*)(Xp + off);
#pragma unroll
      for (int i = 0; i < 4; ++i) r.f[i] = gp[i];
      __builtin_prefetch(Xp + off + 64, 0, 0);  // next-next chunk -> global_prefetch
    }
  } else if (rv) {
    float tmp[16];
#pragma unroll
    for (int c = 0; c < 16; ++c) {
      const int kk = kc + k0 + c;
      float v = 0.0f;
      if (kk < K) v = (kk < K1) ? X1[(long)rg * K1 + kk]
                                : X2[(long)rg * K2 + (kk - K1)];
      tmp[c] = v;
    }
#pragma unroll
    for (int i = 0; i < 4; ++i)
      r.f[i] = make_float4(tmp[4 * i], tmp[4 * i + 1], tmp[4 * i + 2], tmp[4 * i + 3]);
  }
}

__device__ __forceinline__ void store_x_chunk(_Float16* Xs, int row, int k0, const XRegs& r) {
  *(v8h*)&Xs[row * 32 + k0]     = cvt8(r.f[0], r.f[1]);
  *(v8h*)&Xs[row * 32 + k0 + 8] = cvt8(r.f[2], r.f[3]);
}

__device__ __forceinline__ void load_w_chunk(float4& a, float4& b,
    const float* __restrict__ W1, int K, int kc, int tid) {
  a = make_float4(0.f, 0.f, 0.f, 0.f); b = a;
  const int k  = tid >> 3;
  const int kk = kc + k;
  if (kk < K) {
    const int n0 = (tid & 7) * 8;
    const float4* gp = (const float4*)(W1 + (long)kk * 64 + n0);
    a = gp[0]; b = gp[1];
  }
}

__device__ __forceinline__ void store_w_chunk(_Float16* W1t, const float4 a, const float4 b, int tid) {
  const int k  = tid >> 3;
  const int n0 = (tid & 7) * 8;
  W1t[(n0 + 0) * 32 + k] = (_Float16)a.x;
  W1t[(n0 + 1) * 32 + k] = (_Float16)a.y;
  W1t[(n0 + 2) * 32 + k] = (_Float16)a.z;
  W1t[(n0 + 3) * 32 + k] = (_Float16)a.w;
  W1t[(n0 + 4) * 32 + k] = (_Float16)b.x;
  W1t[(n0 + 5) * 32 + k] = (_Float16)b.y;
  W1t[(n0 + 6) * 32 + k] = (_Float16)b.z;
  W1t[(n0 + 7) * 32 + k] = (_Float16)b.w;
}

// ---------------------------------------------------------------------------
// Y = [relu?]( relu( [X1 | X2] @ W1 + b1 ) @ W2 + b2 )
// 256 threads = 8 wave32 waves; block computes 128 rows x 64 cols.
// K loop software-pipelined: global loads for chunk k+32 issued before the
// WMMAs of chunk k; double-buffered LDS; one barrier per chunk.
// ---------------------------------------------------------------------------
__global__ __launch_bounds__(256)
void mlp2_kernel(const float* __restrict__ X1, int K1,
                 const float* __restrict__ X2, int K2,
                 int M,
                 const float* __restrict__ W1, const float* __restrict__ b1,
                 const float* __restrict__ W2, const float* __restrict__ b2,
                 float* __restrict__ Y, int ldY, int finalRelu)
{
  __shared__ __align__(16) _Float16 Xs[2][128 * 32];
  __shared__ __align__(16) _Float16 W1t[2][64 * 32];
  __shared__ __align__(16) _Float16 W2t[64 * 64];
  __shared__ __align__(16) _Float16 Hs[128 * 64];
  __shared__ float b1s[64];
  __shared__ float b2s[64];

  const int tid  = threadIdx.x;
  const int lane = tid & 31;
  const int wave = tid >> 5;
  const int rowBase = blockIdx.x * 128;
  const int K = K1 + K2;

  stage_wt(W2t, 64, W2, 64, tid);
  if (tid < 64) { b1s[tid] = b1[tid]; b2s[tid] = b2[tid]; }

  const int row = tid >> 1;          // 0..127
  const int k0  = (tid & 1) * 16;    // 0 or 16
  const int rg  = rowBase + row;
  const bool rv = (rg < M);

  v8f acc[4];
#pragma unroll
  for (int t = 0; t < 4; ++t) acc[t] = vzero8();

  XRegs rx;
  float4 rwa, rwb;
  load_x_chunk(rx, X1, K1, X2, K2, K, rg, rv, k0, 0);
  load_w_chunk(rwa, rwb, W1, K, 0, tid);

  int buf = 0;
  for (int kc = 0; kc < K; kc += 32) {
    store_x_chunk(Xs[buf], row, k0, rx);
    store_w_chunk(W1t[buf], rwa, rwb, tid);
    __syncthreads();
    if (kc + 32 < K) {
      load_x_chunk(rx, X1, K1, X2, K2, K, rg, rv, k0, kc + 32);
      load_w_chunk(rwa, rwb, W1, K, kc + 32, tid);
    }
    wmma_row(&Xs[buf][wave * 16 * 32], 32, 0, W1t[buf], 32, lane, acc);
    buf ^= 1;
  }

  __syncthreads();
  // bias + relu -> Hs (f16)
  {
    const int rs = (lane >> 4) << 3;
#pragma unroll
    for (int nt = 0; nt < 4; ++nt) {
      const int col = nt * 16 + (lane & 15);
#pragma unroll
      for (int r = 0; r < 8; ++r) {
        const int rr = wave * 16 + r + rs;
        float v = acc[nt][r] + b1s[col];
        v = v > 0.0f ? v : 0.0f;
        Hs[rr * 64 + col] = (_Float16)v;
      }
    }
  }
  __syncthreads();

  // second layer: K = 64
  v8f acc2[4];
#pragma unroll
  for (int t = 0; t < 4; ++t) acc2[t] = vzero8();
#pragma unroll
  for (int kc = 0; kc < 64; kc += 32)
    wmma_row(&Hs[wave * 16 * 64], 64, kc, W2t, 64, lane, acc2);

  // epilogue
  {
    const int rs = (lane >> 4) << 3;
#pragma unroll
    for (int nt = 0; nt < 4; ++nt) {
      const int col = nt * 16 + (lane & 15);
#pragma unroll
      for (int r = 0; r < 8; ++r) {
        const int rr = wave * 16 + r + rs;
        const int rgo = rowBase + rr;
        float v = acc2[nt][r] + b2s[col];
        if (finalRelu) v = v > 0.0f ? v : 0.0f;
        if (rgo < M) Y[(long)rgo * ldY + col] = v;
      }
    }
  }
}

// ---------------------------------------------------------------------------
// Fused GNN edge pipeline for one layer:
//   pi = relu(x[dst] @ Wpi + bpi), pj = relu(x[src] @ Wpj + bpj)
//   msg = relu([pi|pj] @ Wm1 + bm1) @ Wm2 + bm2 ; aggr[dst] += msg (atomics)
// Block = 128 edges; all intermediates stay in LDS (f16).
// ---------------------------------------------------------------------------
__global__ __launch_bounds__(256)
void gnn_edge_kernel(const float* __restrict__ x,
                     const int* __restrict__ src, const int* __restrict__ dst,
                     const float* __restrict__ Wpi, const float* __restrict__ bpi,
                     const float* __restrict__ Wpj, const float* __restrict__ bpj,
                     const float* __restrict__ Wm1, const float* __restrict__ bm1,
                     const float* __restrict__ Wm2, const float* __restrict__ bm2,
                     float* __restrict__ aggr, int E)
{
  __shared__ __align__(16) _Float16 Xg[128 * 64];     // gathered x rows (f16)
  __shared__ __align__(16) _Float16 WptI[64 * 64];    // Wpi^T
  __shared__ __align__(16) _Float16 WptJ[64 * 64];    // Wpj^T
  __shared__ __align__(16) _Float16 PP[128 * 128];    // [pi | pj]
  __shared__ __align__(16) _Float16 Wm1t[64 * 128];   // Wm1^T
  __shared__ __align__(16) _Float16 Wm2t[64 * 64];    // Wm2^T
  __shared__ __align__(16) _Float16 Hs[128 * 64];     // msg hidden
  __shared__ float bpsI[64], bpsJ[64], bm1s[64], bm2s[64];
  __shared__ int scat[128];

  const int tid   = threadIdx.x;
  const int lane  = tid & 31;
  const int wave  = tid >> 5;
  const int eBase = blockIdx.x * 128;

  stage_wt(Wm1t, 128, Wm1, 128, tid);
  stage_wt(Wm2t, 64, Wm2, 64, tid);
  stage_wt(WptI, 64, Wpi, 64, tid);
  stage_wt(WptJ, 64, Wpj, 64, tid);
  if (tid < 64) {
    bpsI[tid] = bpi[tid]; bpsJ[tid] = bpj[tid];
    bm1s[tid] = bm1[tid]; bm2s[tid] = bm2[tid];
  }
  if (tid < 128) {
    const int eg = eBase + tid;
    scat[tid] = (eg < E) ? dst[eg] : -1;
  }

  v8f acc[4];

  // two projection passes: p=0 -> pi from x[dst], p=1 -> pj from x[src]
  for (int p = 0; p < 2; ++p) {
    const int* gidx = (p == 0) ? dst : src;
    __syncthreads();
    // gather x rows -> Xg (f16); each thread: half a node row (32 floats)
    {
      const int row = tid >> 1;
      const int c0  = (tid & 1) * 32;
      const int eg  = eBase + row;
      const bool rvv = (eg < E);
      float4 f[8];
#pragma unroll
      for (int i = 0; i < 8; ++i) f[i] = make_float4(0.f, 0.f, 0.f, 0.f);
      if (rvv) {
        const int node = gidx[eg];
        const float4* gp = (const float4*)(x + (long)node * 64 + c0);
#pragma unroll
        for (int i = 0; i < 8; ++i) f[i] = gp[i];
      }
#pragma unroll
      for (int i = 0; i < 4; ++i)
        *(v8h*)&Xg[row * 64 + c0 + i * 8] = cvt8(f[2 * i], f[2 * i + 1]);
    }
    __syncthreads();

    const _Float16* Wpt = (p == 0) ? WptI : WptJ;
    const float*    bps = (p == 0) ? bpsI : bpsJ;
#pragma unroll
    for (int t = 0; t < 4; ++t) acc[t] = vzero8();
#pragma unroll
    for (int kc = 0; kc < 64; kc += 32)
      wmma_row(&Xg[wave * 16 * 64], 64, kc, Wpt, 64, lane, acc);
    {
      const int rs = (lane >> 4) << 3;
#pragma unroll
      for (int nt = 0; nt < 4; ++nt) {
        const int col = nt * 16 + (lane & 15);
#pragma unroll
        for (int r = 0; r < 8; ++r) {
          const int rr = wave * 16 + r + rs;
          float v = acc[nt][r] + bps[col];
          v = v > 0.0f ? v : 0.0f;
          PP[rr * 128 + p * 64 + col] = (_Float16)v;
        }
      }
    }
  }
  __syncthreads();

  // message layer 1: [128 x 128] @ Wm1 -> 64, relu
#pragma unroll
  for (int t = 0; t < 4; ++t) acc[t] = vzero8();
#pragma unroll
  for (int kc = 0; kc < 128; kc += 32)
    wmma_row(&PP[wave * 16 * 128], 128, kc, Wm1t, 128, lane, acc);
  {
    const int rs = (lane >> 4) << 3;
#pragma unroll
    for (int nt = 0; nt < 4; ++nt) {
      const int col = nt * 16 + (lane & 15);
#pragma unroll
      for (int r = 0; r < 8; ++r) {
        const int rr = wave * 16 + r + rs;
        float v = acc[nt][r] + bm1s[col];
        v = v > 0.0f ? v : 0.0f;
        Hs[rr * 64 + col] = (_Float16)v;
      }
    }
  }
  __syncthreads();

  // message layer 2 + atomic scatter to aggr[dst]
#pragma unroll
  for (int t = 0; t < 4; ++t) acc[t] = vzero8();
#pragma unroll
  for (int kc = 0; kc < 64; kc += 32)
    wmma_row(&Hs[wave * 16 * 64], 64, kc, Wm2t, 64, lane, acc);
  {
    const int rs = (lane >> 4) << 3;
#pragma unroll
    for (int nt = 0; nt < 4; ++nt) {
      const int col = nt * 16 + (lane & 15);
#pragma unroll
      for (int r = 0; r < 8; ++r) {
        const int rr = wave * 16 + r + rs;
        const int d = scat[rr];
        const float v = acc[nt][r] + bm2s[col];
        if (d >= 0) atomicAdd(&aggr[(long)d * 64 + col], v);
      }
    }
  }
}

__global__ void fill_zero_kernel(float* __restrict__ p, long n) {
  const long i = (long)blockIdx.x * blockDim.x + threadIdx.x;
  if (i < n) p[i] = 0.0f;
}

extern "C" void kernel_launch(void* const* d_in, const int* in_sizes, int n_in,
                              void* d_out, int out_size, void* d_ws, size_t ws_size,
                              hipStream_t stream)
{
  (void)in_sizes; (void)n_in; (void)out_size; (void)ws_size;

  const float* rna        = (const float*)d_in[0];
  const float* atac       = (const float*)d_in[1];
  const float* protein    = (const float*)d_in[2];
  const int*   edge_index = (const int*)d_in[3];
  const int*   src = edge_index;            // edge_index[0] = x_j source
  const int*   dst = edge_index + NE;       // edge_index[1] = target / aggregate

  const float* rna_W1 = (const float*)d_in[4];
  const float* rna_b1 = (const float*)d_in[5];
  const float* rna_W2 = (const float*)d_in[6];
  const float* rna_b2 = (const float*)d_in[7];
  const float* atac_W1 = (const float*)d_in[8];
  const float* atac_b1 = (const float*)d_in[9];
  const float* atac_W2 = (const float*)d_in[10];
  const float* atac_b2 = (const float*)d_in[11];
  const float* prot_W1 = (const float*)d_in[12];
  const float* prot_b1 = (const float*)d_in[13];
  const float* prot_W2 = (const float*)d_in[14];
  const float* prot_b2 = (const float*)d_in[15];
  const float* int_W1 = (const float*)d_in[16];
  const float* int_b1 = (const float*)d_in[17];
  const float* int_W2 = (const float*)d_in[18];
  const float* int_b2 = (const float*)d_in[19];
  const float* Wpi = (const float*)d_in[20];
  const float* bpi = (const float*)d_in[21];
  const float* Wpj = (const float*)d_in[22];
  const float* bpj = (const float*)d_in[23];
  const float* Wm1 = (const float*)d_in[24];
  const float* bm1 = (const float*)d_in[25];
  const float* Wm2 = (const float*)d_in[26];
  const float* bm2 = (const float*)d_in[27];
  const float* Wu1 = (const float*)d_in[28];
  const float* bu1 = (const float*)d_in[29];
  const float* Wu2 = (const float*)d_in[30];
  const float* bu2 = (const float*)d_in[31];

  // workspace layout
  float* xc   = (float*)d_ws;                      // [N,192] concat encodings
  float* xA   = xc + (size_t)NN * 192;             // [N,64]
  float* xB   = xA + (size_t)NN * 64;              // [N,64]
  float* aggr = xB + (size_t)NN * 64;              // [N,64]

  const dim3 blk(256);
  const int gridN = (NN + 127) / 128;
  const int gridE = (NE + 127) / 128;

  // modality encoders -> xc columns [0,64), [64,128), [128,192)
  mlp2_kernel<<<gridN, blk, 0, stream>>>(rna, 2000, nullptr, 0, NN,
      rna_W1, rna_b1, rna_W2, rna_b2, xc + 0, 192, 0);
  mlp2_kernel<<<gridN, blk, 0, stream>>>(atac, 1000, nullptr, 0, NN,
      atac_W1, atac_b1, atac_W2, atac_b2, xc + 64, 192, 0);
  mlp2_kernel<<<gridN, blk, 0, stream>>>(protein, 200, nullptr, 0, NN,
      prot_W1, prot_b1, prot_W2, prot_b2, xc + 128, 192, 0);
  // integration MLP -> x
  mlp2_kernel<<<gridN, blk, 0, stream>>>(xc, 192, nullptr, 0, NN,
      int_W1, int_b1, int_W2, int_b2, xA, 64, 0);

  float* xin  = xA;
  float* xout = xB;
  for (int l = 0; l < 2; ++l) {
    const long aggrN = (long)NN * 64;
    fill_zero_kernel<<<(int)((aggrN + 255) / 256), blk, 0, stream>>>(aggr, aggrN);
    gnn_edge_kernel<<<gridE, blk, 0, stream>>>(xin, src, dst,
        Wpi + (size_t)l * 64 * 64, bpi + (size_t)l * 64,
        Wpj + (size_t)l * 64 * 64, bpj + (size_t)l * 64,
        Wm1 + (size_t)l * 128 * 64, bm1 + (size_t)l * 64,
        Wm2 + (size_t)l * 64 * 64, bm2 + (size_t)l * 64,
        aggr, NE);
    float* Yl = (l == 1) ? (float*)d_out : xout;
    // x = relu(mlp2([aggr | x]))
    mlp2_kernel<<<gridN, blk, 0, stream>>>(aggr, 64, xin, 64, NN,
        Wu1 + (size_t)l * 128 * 64, bu1 + (size_t)l * 64,
        Wu2 + (size_t)l * 64 * 64, bu2 + (size_t)l * 64,
        Yl, 64, 1);
    float* t = xin; xin = xout; xout = t;
  }
}
